// KNNAttentionBlock_85993835201279
// MI455X (gfx1250) — compile-verified, matrix operand order checked by reading, not run on hardware
//
#include <hip/hip_runtime.h>
#include <hip/hip_bf16.h>
#include <math.h>

#define NPTS   8192
#define DMODEL 256
#define KNBR   32
#define HPOS   128
#define FFDIM  1024
#define NHEADS 8
#define HDIM   32

typedef __attribute__((ext_vector_type(16))) _Float16 v16h;
typedef __attribute__((ext_vector_type(8)))  _Float16 v8h;
typedef __attribute__((ext_vector_type(8)))  float    v8f;

// ---------------------------------------------------------------------------
// WMMA helpers (CDNA5 v_wmma_f32_16x16x32_f16, wave32)
// A fragment (16x32 f16, row-major src): lane L<16 -> row L, K {0..7,16..23};
// lane L>=16 -> row L-16, K {8..15,24..31}.  B fragment for Y = X @ W^T loads
// W rows (N-index) with the identical pattern.
// ---------------------------------------------------------------------------
__device__ __forceinline__ v16h load_frag(const _Float16* __restrict__ p0, int ld, int lane) {
  const _Float16* p = p0 + (size_t)(lane & 15) * (size_t)ld + (size_t)((lane >> 4) << 3);
  v8h lo = *(const v8h*)(p);
  v8h hi = *(const v8h*)(p + 16);
  return __builtin_shufflevector(lo, hi, 0,1,2,3,4,5,6,7,8,9,10,11,12,13,14,15);
}

__device__ __forceinline__ v8f wmma_f16(v16h a, v16h b, v8f c) {
  return __builtin_amdgcn_wmma_f32_16x16x32_f16(false, a, false, b, (short)0, c, false, false);
}

__device__ __forceinline__ float wave_sum(float v) {
#pragma unroll
  for (int m = 16; m >= 1; m >>= 1) v += __shfl_xor(v, m, 32);
  return v;
}
__device__ __forceinline__ float wave_max(float v) {
#pragma unroll
  for (int m = 16; m >= 1; m >>= 1) v = fmaxf(v, __shfl_xor(v, m, 32));
  return v;
}

// ---------------------------------------------------------------------------
// Kernel 0: f32 -> f16 conversion (weights into workspace)
// ---------------------------------------------------------------------------
__global__ void cvt_f32_f16_kernel(const float* __restrict__ src, _Float16* __restrict__ dst, int n) {
  int i = blockIdx.x * blockDim.x + threadIdx.x;
  if (i < n) dst[i] = (_Float16)src[i];
}

// ---------------------------------------------------------------------------
// Kernel 1: brute-force KNN (top-32 by squared distance, self included)
// ---------------------------------------------------------------------------
__global__ __launch_bounds__(256) void knn_kernel(const float* __restrict__ xyz,
                                                  int* __restrict__ idx_out) {
  __shared__ float sx[256], sy[256], sz[256];
  const int q = blockIdx.x * 256 + threadIdx.x;
  const float qx = xyz[q * 3 + 0], qy = xyz[q * 3 + 1], qz = xyz[q * 3 + 2];
  float bestd[KNBR];
  int   besti[KNBR];
#pragma unroll
  for (int i = 0; i < KNBR; ++i) { bestd[i] = 3.4e38f; besti[i] = 0; }
  for (int t = 0; t < NPTS; t += 256) {
    __syncthreads();
    const int j = t + threadIdx.x;
    sx[threadIdx.x] = xyz[j * 3 + 0];
    sy[threadIdx.x] = xyz[j * 3 + 1];
    sz[threadIdx.x] = xyz[j * 3 + 2];
    __syncthreads();
    for (int jj = 0; jj < 256; ++jj) {
      const float dx = qx - sx[jj], dy = qy - sy[jj], dz = qz - sz[jj];
      const float d = dx * dx + dy * dy + dz * dz;
      if (d < bestd[KNBR - 1]) {
        int p = KNBR - 1;
        while (p > 0 && bestd[p - 1] > d) {
          bestd[p] = bestd[p - 1]; besti[p] = besti[p - 1]; --p;
        }
        bestd[p] = d; besti[p] = t + jj;
      }
    }
  }
#pragma unroll
  for (int i = 0; i < KNBR; ++i) idx_out[(size_t)q * KNBR + i] = besti[i];
}

// ---------------------------------------------------------------------------
// Kernel 2: per-point fused pos-MLP + QKV projections + attention -> ctx (f16)
// One block (8 waves) per point.  Static LDS ~59 KB.
// ---------------------------------------------------------------------------
__global__ __launch_bounds__(256) void point_attn_kernel(
    const float* __restrict__ feats, const float* __restrict__ xyz,
    const int* __restrict__ knn,
    const _Float16* __restrict__ wq_h, const _Float16* __restrict__ wk_h,
    const _Float16* __restrict__ wv_h, const float* __restrict__ in_proj_b,
    const float* __restrict__ pos_w1, const float* __restrict__ pos_b1,
    const float* __restrict__ pos_ln_g, const float* __restrict__ pos_ln_b,
    const _Float16* __restrict__ posw2_h, const float* __restrict__ pos_b2,
    _Float16* __restrict__ ctx_out)
{
  __shared__ __align__(16) _Float16 feats_h[KNBR][DMODEL];  // neighbor feats
  __shared__ __align__(16) _Float16 kin_h[KNBR][DMODEL];    // pos -> feats+pos -> vh
  __shared__ __align__(16) _Float16 ph_h[KNBR][HPOS];       // relu(LN(pos layer1))
  __shared__ __align__(16) _Float16 kh_h[KNBR][DMODEL];     // projected keys
  __shared__ float relv[KNBR][4];
  __shared__ float maskf[KNBR];
  __shared__ int   sidx[KNBR];
  __shared__ float qvec[DMODEL];
  __shared__ float qh[DMODEL];

  const int n = blockIdx.x;
  const int tid = threadIdx.x, lane = tid & 31, wid = tid >> 5;

  // --- neighbor metadata ---
  if (tid < KNBR) {
    const int j = knn[(size_t)n * KNBR + tid];
    sidx[tid] = j;
    const float rx = xyz[j * 3 + 0] - xyz[n * 3 + 0];
    const float ry = xyz[j * 3 + 1] - xyz[n * 3 + 1];
    const float rz = xyz[j * 3 + 2] - xyz[n * 3 + 2];
    relv[tid][0] = rx; relv[tid][1] = ry; relv[tid][2] = rz; relv[tid][3] = 0.f;
    maskf[tid] = (sqrtf(rx * rx + ry * ry + rz * rz) > 0.3f) ? 1.f : 0.f;
  }
  __syncthreads();

  // --- gather neighbor features (f32 -> f16), coalesced per row ---
  for (int i = 0; i < KNBR; ++i)
    feats_h[i][tid] = (_Float16)feats[(size_t)sidx[i] * DMODEL + tid];
  __syncthreads();

  // --- pos MLP layer 1 + LN + ReLU (wave per 4 rows) ---
  for (int nb = wid; nb < KNBR; nb += 8) {
    const float rx = relv[nb][0] * 0.1f, ry = relv[nb][1] * 0.1f, rz = relv[nb][2] * 0.1f;
    float raw[4]; float s = 0.f, s2 = 0.f;
#pragma unroll
    for (int j = 0; j < 4; ++j) {
      const int h = lane + 32 * j;
      const float r = rx * pos_w1[h * 3 + 0] + ry * pos_w1[h * 3 + 1] +
                      rz * pos_w1[h * 3 + 2] + pos_b1[h];
      raw[j] = r; s += r; s2 += r * r;
    }
    s = wave_sum(s); s2 = wave_sum(s2);
    const float m = s * (1.f / HPOS);
    const float inv = rsqrtf(s2 * (1.f / HPOS) - m * m + 1e-5f);
#pragma unroll
    for (int j = 0; j < 4; ++j) {
      const int h = lane + 32 * j;
      const float v = (raw[j] - m) * inv * pos_ln_g[h] + pos_ln_b[h];
      ph_h[nb][h] = (_Float16)fmaxf(v, 0.f);
    }
  }
  __syncthreads();

  // --- pos MLP layer 2 (WMMA): pos(32x256) = ph(32x128) @ pos_w2^T + b2 -> kin_h ---
  for (int t = wid; t < 32; t += 8) {
    const int mt = t >> 4, nt = t & 15;
    v8f c = {};
    const _Float16* A = &ph_h[mt * 16][0];
    const _Float16* B = posw2_h + (size_t)(nt * 16) * HPOS;
    for (int kb = 0; kb < HPOS; kb += 32)
      c = wmma_f16(load_frag(A + kb, HPOS, lane), load_frag(B + kb, HPOS, lane), c);
    const int col = lane & 15, rb = (lane >> 4) << 3;
#pragma unroll
    for (int r = 0; r < 8; ++r) {
      const int row = mt * 16 + rb + r, cc = nt * 16 + col;
      kin_h[row][cc] = (_Float16)(c[r] + pos_b2[cc]);
    }
  }
  __syncthreads();

  // --- q input = src + pos[0] ---
  qvec[tid] = feats[(size_t)n * DMODEL + tid] + (float)kin_h[0][tid];
  __syncthreads();

  // --- kin = feats + pos (in place) ---
  for (int i = 0; i < KNBR; ++i)
    kin_h[i][tid] = (_Float16)((float)kin_h[i][tid] + (float)feats_h[i][tid]);
  __syncthreads();

  // --- qh = q @ wq^T + bq (per-thread dot, M=1) ---
  {
    float acc = in_proj_b[tid];
    const _Float16* wr = wq_h + (size_t)tid * DMODEL;
    for (int d = 0; d < DMODEL; ++d) acc += qvec[d] * (float)wr[d];
    qh[tid] = acc;
  }

  // --- key projection (WMMA): kh = kin @ wk^T + bk ---
  for (int t = wid; t < 32; t += 8) {
    const int mt = t >> 4, nt = t & 15;
    v8f c = {};
    const _Float16* A = &kin_h[mt * 16][0];
    const _Float16* B = wk_h + (size_t)(nt * 16) * DMODEL;
    for (int kb = 0; kb < DMODEL; kb += 32)
      c = wmma_f16(load_frag(A + kb, DMODEL, lane), load_frag(B + kb, DMODEL, lane), c);
    const int col = lane & 15, rb = (lane >> 4) << 3;
#pragma unroll
    for (int r = 0; r < 8; ++r) {
      const int row = mt * 16 + rb + r, cc = nt * 16 + col;
      kh_h[row][cc] = (_Float16)(c[r] + in_proj_b[DMODEL + cc]);
    }
  }
  __syncthreads();

  // --- value projection (WMMA): vh = feats @ wv^T + bv  (overwrites kin_h) ---
  for (int t = wid; t < 32; t += 8) {
    const int mt = t >> 4, nt = t & 15;
    v8f c = {};
    const _Float16* A = &feats_h[mt * 16][0];
    const _Float16* B = wv_h + (size_t)(nt * 16) * DMODEL;
    for (int kb = 0; kb < DMODEL; kb += 32)
      c = wmma_f16(load_frag(A + kb, DMODEL, lane), load_frag(B + kb, DMODEL, lane), c);
    const int col = lane & 15, rb = (lane >> 4) << 3;
#pragma unroll
    for (int r = 0; r < 8; ++r) {
      const int row = mt * 16 + rb + r, cc = nt * 16 + col;
      kin_h[row][cc] = (_Float16)(c[r] + in_proj_b[2 * DMODEL + cc]);
    }
  }
  __syncthreads();

  // --- attention: wave = head, lane = neighbor (scores) then lane = dim (ctx) ---
  {
    const int h = wid, k = lane;
    float s = 0.f;
    for (int d = 0; d < HDIM; ++d)
      s += qh[h * HDIM + d] * (float)kh_h[k][h * HDIM + d];
    s *= 0.17677669529663687f;             // 1/sqrt(32)
    if (maskf[k] > 0.5f) s = -INFINITY;
    const float mx = wave_max(s);
    const float e = __expf(s - mx);
    const float denom = wave_sum(e);
    const float attn = e / denom;
    float ctx = 0.f;
    for (int kk = 0; kk < KNBR; ++kk) {
      const float a = __shfl(attn, kk, 32);
      ctx += a * (float)kin_h[kk][h * HDIM + lane];  // kin_h now holds vh
    }
    ctx_out[(size_t)n * DMODEL + h * HDIM + lane] = (_Float16)ctx;
  }
}

// ---------------------------------------------------------------------------
// Kernel 3: src2 = ctx @ out_w^T + out_b; src = LN1(src_in + src2)
// 32 rows per block, WMMA tiles.
// ---------------------------------------------------------------------------
__global__ __launch_bounds__(256) void outproj_ln_kernel(
    const _Float16* __restrict__ ctx_h, const _Float16* __restrict__ outw_h,
    const float* __restrict__ out_b, const float* __restrict__ src_in,
    const float* __restrict__ ln1_g, const float* __restrict__ ln1_b,
    float* __restrict__ src_f, _Float16* __restrict__ src_h)
{
  __shared__ __align__(16) _Float16 a_h[32][DMODEL];
  __shared__ __align__(16) float    yf[32][DMODEL];
  const int rb0 = blockIdx.x * 32;
  const int tid = threadIdx.x, lane = tid & 31, wid = tid >> 5;

  for (int i = 0; i < 32; ++i)
    a_h[i][tid] = ctx_h[(size_t)(rb0 + i) * DMODEL + tid];
  __syncthreads();

  for (int t = wid; t < 32; t += 8) {
    const int mt = t >> 4, nt = t & 15;
    v8f c = {};
    const _Float16* A = &a_h[mt * 16][0];
    const _Float16* B = outw_h + (size_t)(nt * 16) * DMODEL;
    for (int kb = 0; kb < DMODEL; kb += 32)
      c = wmma_f16(load_frag(A + kb, DMODEL, lane), load_frag(B + kb, DMODEL, lane), c);
    const int col = lane & 15, rr = (lane >> 4) << 3;
#pragma unroll
    for (int r = 0; r < 8; ++r) {
      const int row = mt * 16 + rr + r, cc = nt * 16 + col;
      yf[row][cc] = c[r] + out_b[cc];
    }
  }
  __syncthreads();

  for (int rr = wid; rr < 32; rr += 8) {
    const int grow = rb0 + rr;
    float vals[8]; float s = 0.f, s2 = 0.f;
#pragma unroll
    for (int j = 0; j < 8; ++j) {
      const int d = lane + 32 * j;
      const float v = yf[rr][d] + src_in[(size_t)grow * DMODEL + d];
      vals[j] = v; s += v; s2 += v * v;
    }
    s = wave_sum(s); s2 = wave_sum(s2);
    const float m = s * (1.f / DMODEL);
    const float inv = rsqrtf(s2 * (1.f / DMODEL) - m * m + 1e-5f);
#pragma unroll
    for (int j = 0; j < 8; ++j) {
      const int d = lane + 32 * j;
      const float o = (vals[j] - m) * inv * ln1_g[d] + ln1_b[d];
      src_f[(size_t)grow * DMODEL + d] = o;
      src_h[(size_t)grow * DMODEL + d] = (_Float16)o;
    }
  }
}

// ---------------------------------------------------------------------------
// Kernel 4: fused FFN: out = LN2(src + gelu(src@lin1^T+b1)@lin2^T+b2)
// Hidden split into two 512-column halves to stay under 64 KB static LDS;
// GEMM2 accumulators persist in registers across halves.
// ---------------------------------------------------------------------------
__global__ __launch_bounds__(256) void ffn_ln_kernel(
    const _Float16* __restrict__ src_h, const float* __restrict__ src_f,
    const _Float16* __restrict__ lin1_h, const float* __restrict__ lin1_b,
    const _Float16* __restrict__ lin2_h, const float* __restrict__ lin2_b,
    const float* __restrict__ ln2_g, const float* __restrict__ ln2_b,
    float* __restrict__ out)
{
  __shared__ __align__(16) _Float16 a_h[32][DMODEL];   // 16 KB
  __shared__ __align__(16) float    hbuf[32 * DMODEL]; // 32 KB: hidden f16 then yf f32
  _Float16* h_h = (_Float16*)hbuf;                     // viewed as [32][512] f16

  const int rb0 = blockIdx.x * 32;
  const int tid = threadIdx.x, lane = tid & 31, wid = tid >> 5;

  for (int i = 0; i < 32; ++i)
    a_h[i][tid] = src_h[(size_t)(rb0 + i) * DMODEL + tid];
  __syncthreads();

  v8f acc[4] = {{}, {}, {}, {}};

  for (int half = 0; half < 2; ++half) {
    // GEMM1 + exact GELU: hidden cols [half*512, half*512+512)
    for (int t = wid; t < 64; t += 8) {
      const int mt = t >> 5, nt = t & 31;
      v8f c = {};
      const _Float16* A = &a_h[mt * 16][0];
      const _Float16* B = lin1_h + (size_t)(half * 512 + nt * 16) * DMODEL;
      for (int kb = 0; kb < DMODEL; kb += 32)
        c = wmma_f16(load_frag(A + kb, DMODEL, lane), load_frag(B + kb, DMODEL, lane), c);
      const int col = lane & 15, rr = (lane >> 4) << 3;
#pragma unroll
      for (int r = 0; r < 8; ++r) {
        const int row = mt * 16 + rr + r, cc = nt * 16 + col;
        const float x = c[r] + lin1_b[half * 512 + cc];
        const float g = 0.5f * x * (1.f + erff(x * 0.70710678118f));
        h_h[(size_t)row * 512 + cc] = (_Float16)g;
      }
    }
    __syncthreads();

    // GEMM2 partial accumulation over this half's 512 K values
#pragma unroll
    for (int ti = 0; ti < 4; ++ti) {
      const int t = wid * 4 + ti;
      const int mt = t >> 4, nt = t & 15;
      const _Float16* A = h_h + (size_t)(mt * 16) * 512;
      const _Float16* B = lin2_h + (size_t)(nt * 16) * FFDIM + half * 512;
      v8f c = acc[ti];
      for (int kb = 0; kb < 512; kb += 32)
        c = wmma_f16(load_frag(A + kb, 512, lane), load_frag(B + kb, FFDIM, lane), c);
      acc[ti] = c;
    }
    __syncthreads();  // before next half overwrites h_h
  }

  // write yf (reuse hbuf as f32 [32][256]) + bias
  float* yf = hbuf;
#pragma unroll
  for (int ti = 0; ti < 4; ++ti) {
    const int t = wid * 4 + ti;
    const int mt = t >> 4, nt = t & 15;
    const int col = lane & 15, rr = (lane >> 4) << 3;
#pragma unroll
    for (int r = 0; r < 8; ++r) {
      const int row = mt * 16 + rr + r, cc = nt * 16 + col;
      yf[(size_t)row * DMODEL + cc] = acc[ti][r] + lin2_b[cc];
    }
  }
  __syncthreads();

  // residual + LN2 -> out
  for (int rr = wid; rr < 32; rr += 8) {
    const int grow = rb0 + rr;
    float vals[8]; float s = 0.f, s2 = 0.f;
#pragma unroll
    for (int j = 0; j < 8; ++j) {
      const int d = lane + 32 * j;
      const float v = yf[(size_t)rr * DMODEL + d] + src_f[(size_t)grow * DMODEL + d];
      vals[j] = v; s += v; s2 += v * v;
    }
    s = wave_sum(s); s2 = wave_sum(s2);
    const float m = s * (1.f / DMODEL);
    const float inv = rsqrtf(s2 * (1.f / DMODEL) - m * m + 1e-5f);
#pragma unroll
    for (int j = 0; j < 8; ++j) {
      const int d = lane + 32 * j;
      out[(size_t)grow * DMODEL + d] = (vals[j] - m) * inv * ln2_g[d] + ln2_b[d];
    }
  }
}

// ---------------------------------------------------------------------------
// Host launch
// ---------------------------------------------------------------------------
extern "C" void kernel_launch(void* const* d_in, const int* in_sizes, int n_in,
                              void* d_out, int out_size, void* d_ws, size_t ws_size,
                              hipStream_t stream) {
  const float* pts_feats = (const float*)d_in[0];
  const float* pts_xyz   = (const float*)d_in[1];
  // d_in[2] = pts_inds (unused)
  const float* in_proj_w = (const float*)d_in[3];
  const float* in_proj_b = (const float*)d_in[4];
  const float* out_w     = (const float*)d_in[5];
  const float* out_b     = (const float*)d_in[6];
  const float* pos_w1    = (const float*)d_in[7];
  const float* pos_b1    = (const float*)d_in[8];
  const float* pos_ln_g  = (const float*)d_in[9];
  const float* pos_ln_b  = (const float*)d_in[10];
  const float* pos_w2    = (const float*)d_in[11];
  const float* pos_b2    = (const float*)d_in[12];
  const float* lin1_w    = (const float*)d_in[13];
  const float* lin1_b    = (const float*)d_in[14];
  const float* lin2_w    = (const float*)d_in[15];
  const float* lin2_b    = (const float*)d_in[16];
  const float* ln1_g     = (const float*)d_in[17];
  const float* ln1_b     = (const float*)d_in[18];
  const float* ln2_g     = (const float*)d_in[19];
  const float* ln2_b     = (const float*)d_in[20];
  float* outp = (float*)d_out;

  // workspace carve-up (256-byte aligned)
  size_t off = 0;
  auto take = [&](size_t bytes) {
    size_t r = off;
    off += (bytes + 255) & ~(size_t)255;
    return r;
  };
  char* ws = (char*)d_ws;
  int*       idx_ws  = (int*)      (ws + take((size_t)NPTS * KNBR * 4));
  _Float16*  wq_h    = (_Float16*) (ws + take((size_t)DMODEL * DMODEL * 2));
  _Float16*  wk_h    = (_Float16*) (ws + take((size_t)DMODEL * DMODEL * 2));
  _Float16*  wv_h    = (_Float16*) (ws + take((size_t)DMODEL * DMODEL * 2));
  _Float16*  outw_h  = (_Float16*) (ws + take((size_t)DMODEL * DMODEL * 2));
  _Float16*  posw2_h = (_Float16*) (ws + take((size_t)DMODEL * HPOS * 2));
  _Float16*  lin1_h  = (_Float16*) (ws + take((size_t)FFDIM * DMODEL * 2));
  _Float16*  lin2_h  = (_Float16*) (ws + take((size_t)DMODEL * FFDIM * 2));
  _Float16*  ctx_h   = (_Float16*) (ws + take((size_t)NPTS * DMODEL * 2));
  float*     src_f   = (float*)    (ws + take((size_t)NPTS * DMODEL * 4));
  _Float16*  src_h   = (_Float16*) (ws + take((size_t)NPTS * DMODEL * 2));
  (void)ws_size; (void)in_sizes; (void)n_in; (void)out_size;

  auto cvt = [&](const float* s, _Float16* d, int n) {
    cvt_f32_f16_kernel<<<(n + 255) / 256, 256, 0, stream>>>(s, d, n);
  };
  cvt(in_proj_w,                     wq_h,    DMODEL * DMODEL);
  cvt(in_proj_w + DMODEL * DMODEL,   wk_h,    DMODEL * DMODEL);
  cvt(in_proj_w + 2 * DMODEL*DMODEL, wv_h,    DMODEL * DMODEL);
  cvt(out_w,                         outw_h,  DMODEL * DMODEL);
  cvt(pos_w2,                        posw2_h, DMODEL * HPOS);
  cvt(lin1_w,                        lin1_h,  FFDIM * DMODEL);
  cvt(lin2_w,                        lin2_h,  DMODEL * FFDIM);

  knn_kernel<<<NPTS / 256, 256, 0, stream>>>(pts_xyz, idx_ws);

  point_attn_kernel<<<NPTS, 256, 0, stream>>>(
      pts_feats, pts_xyz, idx_ws, wq_h, wk_h, wv_h, in_proj_b,
      pos_w1, pos_b1, pos_ln_g, pos_ln_b, posw2_h, pos_b2, ctx_h);

  outproj_ln_kernel<<<NPTS / 32, 256, 0, stream>>>(
      ctx_h, outw_h, out_b, pts_feats, ln1_g, ln1_b, src_f, src_h);

  ffn_ln_kernel<<<NPTS / 32, 256, 0, stream>>>(
      src_h, src_f, lin1_h, lin1_b, lin2_h, lin2_b, ln2_g, ln2_b, outp);
}
